// LayerNormNeuralCDE_850403525426
// MI455X (gfx1250) — matrix operand order repaired and checked by reading, not compiled, not run on hardware
//
#include <hip/hip_runtime.h>
#include <hip/hip_bf16.h>
#include <math.h>

typedef __bf16 bf16_t;
typedef __attribute__((ext_vector_type(16))) __bf16 v16bf;
typedef __attribute__((ext_vector_type(8)))  float  v8f;

#define HIDN 256
#define DINN 64
#define BSZ  32
#define SEQ  16
#define NTSK 20
#define H2D  512

// ---------------- fragment loaders ----------------
// A-matrix 16x32 bf16 fragment built from fp32 source, optional fused axpy
// (A + alpha*ADD). Per ISA: lanes 0-15 -> M=lane, K = {kb..kb+7, kb+16..kb+23};
// lanes 16-31 -> M=lane-16, K = {kb+8..kb+15, kb+24..kb+31}.
__device__ __forceinline__ v16bf load_fragA_f32(const float* __restrict__ base,
                                                const float* __restrict__ addv,
                                                float alpha, int ld, int row0, int kb) {
  const int lane = threadIdx.x & 31;
  const int r    = lane & 15;
  const int hi   = lane >> 4;
  const long off = (long)(row0 + r) * ld + kb + hi * 8;
  const float* p = base + off;
  float v[16];
#pragma unroll
  for (int i = 0; i < 8; ++i) v[i] = p[i];
#pragma unroll
  for (int i = 0; i < 8; ++i) v[8 + i] = p[16 + i];
  if (addv) {
    const float* q = addv + off;
#pragma unroll
    for (int i = 0; i < 8; ++i) v[i] += alpha * q[i];
#pragma unroll
    for (int i = 0; i < 8; ++i) v[8 + i] += alpha * q[16 + i];
  }
  v16bf f;
#pragma unroll
  for (int i = 0; i < 16; ++i) f[i] = (__bf16)v[i];
  return f;
}

// A-matrix fragment straight from bf16 activations (no conversion): two b128 loads.
__device__ __forceinline__ v16bf load_fragA_bf16(const bf16_t* __restrict__ base,
                                                 int ld, int row0, int kb) {
  const int lane = threadIdx.x & 31;
  const int r    = lane & 15;
  const int hi   = lane >> 4;
  const bf16_t* p = base + (long)(row0 + r) * ld + kb + hi * 8;
  union { struct { uint4 a; uint4 b; } u; v16bf f; } t;
  t.u.a = *(const uint4*)(p);        // K run 1 (8 bf16 = 16B)
  t.u.b = *(const uint4*)(p + 16);   // K run 2
  return t.f;
}

// B-matrix 32x16 bf16 fragment from pre-transposed weights W^T stored [N][K].
// lanes 0-15 -> N=lane, K={kb..kb+7, kb+16..kb+23}; lanes 16-31 -> N=lane-16, K={kb+8..kb+15, kb+24..kb+31}.
__device__ __forceinline__ v16bf load_fragB(const bf16_t* __restrict__ base,
                                            int ld, int col0, int kb) {
  const int lane = threadIdx.x & 31;
  const int n    = lane & 15;
  const int hi   = lane >> 4;
  const bf16_t* p = base + (long)(col0 + n) * ld + kb + hi * 8;
  union { struct { uint4 a; uint4 b; } u; v16bf f; } t;
  t.u.a = *(const uint4*)(p);
  t.u.b = *(const uint4*)(p + 16);
  return t.f;
}

__device__ __forceinline__ float gelu_exact(float x) {
  return 0.5f * x * (1.0f + erff(x * 0.70710678118654752440f));
}

// ---------------- generic GEMM (+bias, +LN, +GELU) ----------------
// C[32][NDIM] = act(LN(A[32][KDIM] @ W[KDIM][NDIM] + bias))
// A: fp32 (with optional fused A0 + coef*(*dtp)*A1) or bf16, per A_BF16.
// Output fp32 or bf16 per OUT_BF16. BT is bf16 W^T stored [NDIM][KDIM].
// grid = (2, batch). block = 128 (4 waves).
template <int KDIM, int NDIM, bool DO_LN, bool DO_GELU, bool A_BF16, bool OUT_BF16>
__global__ __launch_bounds__(128) void gemm_act_kernel(
    const void* __restrict__ Ain, const float* __restrict__ A1,
    const float* __restrict__ dtp, float coef, int lda, int aStrideB,
    const bf16_t* __restrict__ BT, const float* __restrict__ bias,
    const float* __restrict__ gamma, const float* __restrict__ beta,
    void* __restrict__ out) {
  constexpr int NTILES = NDIM / 16;
  constexpr int TMAX   = (NTILES + 3) / 4;

  __shared__ float rowbuf[16 * NDIM];
  __shared__ float red_s[16 * 8];
  __shared__ float red_q[16 * 8];
  __shared__ float s_mu[16];
  __shared__ float s_rs[16];

  const int wave = threadIdx.x >> 5;
  const int lane = threadIdx.x & 31;
  const int m0   = blockIdx.x * 16;
  const int bt   = blockIdx.y;

  const float*  Af = (const float*)Ain  + (long)bt * aStrideB;
  const bf16_t* Ab = (const bf16_t*)Ain + (long)bt * aStrideB;
  const bf16_t* Bw = BT   + (long)bt * KDIM * NDIM;
  const float*  bb = bias + (long)bt * NDIM;

  float alpha = coef;
  if (!A_BF16 && A1 && dtp) alpha = coef * dtp[0];

  v8f acc[TMAX];
#pragma unroll
  for (int j = 0; j < TMAX; ++j)
#pragma unroll
    for (int r = 0; r < 8; ++r) acc[j][r] = 0.0f;

  for (int kb = 0; kb < KDIM; kb += 32) {
    v16bf af = A_BF16 ? load_fragA_bf16(Ab, lda, m0, kb)
                      : load_fragA_f32(Af, A1, alpha, lda, m0, kb);
    int j = 0;
    for (int nt = wave; nt < NTILES; nt += 4, ++j) {
      v16bf bfg = load_fragB(Bw, KDIM, nt * 16, kb);
      acc[j] = __builtin_amdgcn_wmma_f32_16x16x32_bf16(
          false, af, false, bfg, (short)0, acc[j], false, false);
    }
  }

  // spill accumulators (+bias) to LDS: C layout row = r+8*hi, col = lane&15
  {
    const int n  = lane & 15;
    const int hi = lane >> 4;
    int j = 0;
    for (int nt = wave; nt < NTILES; nt += 4, ++j) {
#pragma unroll
      for (int r = 0; r < 8; ++r) {
        int row = r + 8 * hi;
        int col = nt * 16 + n;
        rowbuf[row * NDIM + col] = acc[j][r] + bb[col];
      }
    }
  }
  __syncthreads();

  if (DO_LN) {
    const int row = threadIdx.x >> 3;  // 16 rows x 8 partial-threads
    const int sub = threadIdx.x & 7;
    float s = 0.f, q = 0.f;
    for (int c = sub; c < NDIM; c += 8) {
      float x = rowbuf[row * NDIM + c];
      s += x; q += x * x;
    }
    red_s[row * 8 + sub] = s;
    red_q[row * 8 + sub] = q;
    __syncthreads();
    if (threadIdx.x < 16) {
      float ss = 0.f, qq = 0.f;
      for (int i = 0; i < 8; ++i) { ss += red_s[threadIdx.x * 8 + i]; qq += red_q[threadIdx.x * 8 + i]; }
      float mu  = ss * (1.0f / NDIM);
      float var = qq * (1.0f / NDIM) - mu * mu;
      s_mu[threadIdx.x] = mu;
      s_rs[threadIdx.x] = rsqrtf(var + 1e-5f);
    }
    __syncthreads();
  }

  for (int idx = threadIdx.x; idx < 16 * NDIM; idx += 128) {
    int row = idx / NDIM, col = idx % NDIM;
    float x = rowbuf[idx];
    if (DO_LN)   x = (x - s_mu[row]) * s_rs[row] * gamma[col] + beta[col];
    if (DO_GELU) x = gelu_exact(x);
    long oidx = (long)bt * 32 * NDIM + (m0 + row) * NDIM + col;
    if (OUT_BF16) ((bf16_t*)out)[oidx] = (bf16_t)x;
    else          ((float*)out)[oidx]  = x;
  }
}

// ---------------- fused vf_W3 GEMM + einsum('bhi,bi->bh') ----------------
// Each block: one 16-row tile x one h' (64 columns of f). grid=(2,256), block=128.
// h2 is bf16 [32][512]; zero-conversion WMMA feed path.
__global__ __launch_bounds__(128) void vf3_contract_kernel(
    const bf16_t* __restrict__ h2, const bf16_t* __restrict__ W3T,
    const float* __restrict__ b3, const float* __restrict__ dX,
    float* __restrict__ kout) {
  __shared__ float dxs[16 * DINN];
  __shared__ float wpart[4 * 16];

  const int wave = threadIdx.x >> 5;
  const int lane = threadIdx.x & 31;
  const int m0   = blockIdx.x * 16;
  const int hp   = blockIdx.y;              // h' index 0..255
  const int n0   = hp * DINN + wave * 16;   // global column of f

  for (int i = threadIdx.x; i < 16 * DINN; i += 128)
    dxs[i] = dX[(m0 + i / DINN) * DINN + (i % DINN)];

  v8f acc;
#pragma unroll
  for (int r = 0; r < 8; ++r) acc[r] = 0.0f;

  for (int kb = 0; kb < H2D; kb += 32) {
    if (kb + 32 < H2D)
      __builtin_prefetch(W3T + (long)(n0 + (lane & 15)) * H2D + kb + 32, 0, 1);
    v16bf af  = load_fragA_bf16(h2, H2D, m0, kb);
    v16bf bfg = load_fragB(W3T, H2D, n0, kb);
    acc = __builtin_amdgcn_wmma_f32_16x16x32_bf16(
        false, af, false, bfg, (short)0, acc, false, false);
  }
  __syncthreads();  // dxs ready

  const int n  = lane & 15;
  const int hi = lane >> 4;
  const int cw = wave * 16 + n;      // column within 0..63 (== i index)
  const float bv = b3[n0 + n];
  float vals[8];
#pragma unroll
  for (int r = 0; r < 8; ++r) {
    int row = r + 8 * hi;
    vals[r] = (acc[r] + bv) * dxs[row * DINN + cw];
  }
  // deterministic butterfly sum across the 16-lane half (over columns)
#pragma unroll
  for (int r = 0; r < 8; ++r) {
    float v = vals[r];
    v += __shfl_xor(v, 1);
    v += __shfl_xor(v, 2);
    v += __shfl_xor(v, 4);
    v += __shfl_xor(v, 8);
    vals[r] = v;
  }
  if (n == 0) {
#pragma unroll
    for (int r = 0; r < 8; ++r) wpart[wave * 16 + r + 8 * hi] = vals[r];
  }
  __syncthreads();
  if (threadIdx.x < 16) {
    float s = wpart[threadIdx.x] + wpart[16 + threadIdx.x] +
              wpart[32 + threadIdx.x] + wpart[48 + threadIdx.x];
    kout[(m0 + threadIdx.x) * HIDN + hp] = s;
  }
}

// ---------------- small helper kernels ----------------
__global__ void rk4_update_kernel(float* __restrict__ zc,
                                  const float* __restrict__ k1, const float* __restrict__ k2,
                                  const float* __restrict__ k3, const float* __restrict__ k4,
                                  const float* __restrict__ dtp) {
  int i = blockIdx.x * blockDim.x + threadIdx.x;
  if (i < BSZ * HIDN) {
    float h6 = dtp[0] * (1.0f / 6.0f);
    zc[i] += h6 * (k1[i] + 2.0f * k2[i] + 2.0f * k3[i] + k4[i]);
  }
}

__global__ void prep_dx_kernel(const float* __restrict__ path, const float* __restrict__ ts,
                               float* __restrict__ dts, float* __restrict__ dX) {
  int idx = blockIdx.x * blockDim.x + threadIdx.x;
  if (idx < SEQ - 1) dts[idx] = ts[idx + 1] - ts[idx];
  int total = (SEQ - 1) * BSZ * DINN;
  if (idx < total) {
    int s = idx / (BSZ * DINN);
    int b = (idx / DINN) % BSZ;
    int i = idx % DINN;
    float dt = ts[s + 1] - ts[s];
    dX[idx] = (path[(b * SEQ + s + 1) * DINN + i] - path[(b * SEQ + s) * DINN + i]) / dt;
  }
}

// fp32 [K][N] (batched) -> bf16 [N][K]
__global__ void convert_transpose_kernel(const float* __restrict__ in,
                                         bf16_t* __restrict__ out, int K, int N) {
  long base = (long)blockIdx.y * K * N;
  int idx = blockIdx.x * blockDim.x + threadIdx.x;
  if (idx < K * N) {
    int nn = idx / K, kk = idx % K;
    out[base + (long)nn * K + kk] = (bf16_t)in[base + (long)kk * N + nn];
  }
}

__global__ void dec_out_kernel(const float* __restrict__ h2, const float* __restrict__ W3,
                               const float* __restrict__ b3, float* __restrict__ out) {
  int idx = blockIdx.x * blockDim.x + threadIdx.x;
  if (idx < NTSK * BSZ) {
    int t = idx / BSZ, b = idx % BSZ;
    float s = b3[t];
    const float* hr = h2 + (t * BSZ + b) * 32;
    const float* w  = W3 + t * 32;
#pragma unroll
    for (int m = 0; m < 32; ++m) s += hr[m] * w[m];
    out[idx] = 1.0f / (1.0f + expf(-s));
  }
}

// ---------------- host orchestration ----------------
extern "C" void kernel_launch(void* const* d_in, const int* in_sizes, int n_in,
                              void* d_out, int out_size, void* d_ws, size_t ws_size,
                              hipStream_t stream) {
  (void)in_sizes; (void)n_in; (void)out_size; (void)ws_size;
  const float* path    = (const float*)d_in[0];
  const float* tstamp  = (const float*)d_in[1];
  const float* enc_W1  = (const float*)d_in[2];
  const float* enc_b1  = (const float*)d_in[3];
  const float* enc_g1  = (const float*)d_in[4];
  const float* enc_be1 = (const float*)d_in[5];
  const float* enc_W2  = (const float*)d_in[6];
  const float* enc_b2  = (const float*)d_in[7];
  const float* enc_g2  = (const float*)d_in[8];
  const float* enc_be2 = (const float*)d_in[9];
  const float* vf_W1   = (const float*)d_in[10];
  const float* vf_b1   = (const float*)d_in[11];
  const float* vf_g1   = (const float*)d_in[12];
  const float* vf_be1  = (const float*)d_in[13];
  const float* vf_W2   = (const float*)d_in[14];
  const float* vf_b2   = (const float*)d_in[15];
  const float* vf_g2   = (const float*)d_in[16];
  const float* vf_be2  = (const float*)d_in[17];
  const float* vf_W3   = (const float*)d_in[18];
  const float* vf_b3   = (const float*)d_in[19];
  const float* attn_Wv = (const float*)d_in[20];
  const float* attn_bv = (const float*)d_in[21];
  const float* attn_Wo = (const float*)d_in[22];
  const float* attn_bo = (const float*)d_in[23];
  const float* dec_W1  = (const float*)d_in[24];
  const float* dec_b1  = (const float*)d_in[25];
  const float* dec_W2  = (const float*)d_in[26];
  const float* dec_b2  = (const float*)d_in[27];
  const float* dec_W3  = (const float*)d_in[28];
  const float* dec_b3  = (const float*)d_in[29];

  // workspace carve-out (deterministic, same every call)
  char* wp = (char*)d_ws;
  auto carve = [&](size_t bytes) -> char* {
    char* p = wp;
    wp += (bytes + 255) & ~(size_t)255;
    return p;
  };
  bf16_t* WT_enc1 = (bf16_t*)carve((size_t)64 * 256 * 2);
  bf16_t* WT_enc2 = (bf16_t*)carve((size_t)256 * 256 * 2);
  bf16_t* WT_vf1  = (bf16_t*)carve((size_t)256 * 512 * 2);
  bf16_t* WT_vf2  = (bf16_t*)carve((size_t)512 * 512 * 2);
  bf16_t* WT_vf3  = (bf16_t*)carve((size_t)512 * 16384 * 2);
  bf16_t* WT_av   = (bf16_t*)carve((size_t)256 * 256 * 2);
  bf16_t* WT_ao   = (bf16_t*)carve((size_t)256 * 256 * 2);
  bf16_t* WT_d1   = (bf16_t*)carve((size_t)NTSK * 256 * 64 * 2);
  bf16_t* WT_d2   = (bf16_t*)carve((size_t)NTSK * 64 * 32 * 2);
  float*  dts  = (float*)carve((SEQ - 1) * 4);
  float*  dX   = (float*)carve((size_t)(SEQ - 1) * BSZ * DINN * 4);
  float*  zc   = (float*)carve((size_t)BSZ * HIDN * 4);
  bf16_t* henc = (bf16_t*)carve((size_t)BSZ * HIDN * 2);
  bf16_t* h1   = (bf16_t*)carve((size_t)BSZ * H2D * 2);
  bf16_t* h2   = (bf16_t*)carve((size_t)BSZ * H2D * 2);
  float*  k1   = (float*)carve((size_t)BSZ * HIDN * 4);
  float*  k2   = (float*)carve((size_t)BSZ * HIDN * 4);
  float*  k3   = (float*)carve((size_t)BSZ * HIDN * 4);
  float*  k4   = (float*)carve((size_t)BSZ * HIDN * 4);
  bf16_t* vbuf = (bf16_t*)carve((size_t)BSZ * HIDN * 2);
  bf16_t* abuf = (bf16_t*)carve((size_t)BSZ * HIDN * 2);
  bf16_t* h1d  = (bf16_t*)carve((size_t)NTSK * BSZ * 64 * 2);
  float*  h2d  = (float*)carve((size_t)NTSK * BSZ * 32 * 4);

  // 1) convert + transpose weights to bf16 [N][K]
  auto tconv = [&](const float* in, bf16_t* out, int K, int N, int batch) {
    dim3 g((K * N + 255) / 256, batch);
    convert_transpose_kernel<<<g, 256, 0, stream>>>(in, out, K, N);
  };
  tconv(enc_W1, WT_enc1, 64, 256, 1);
  tconv(enc_W2, WT_enc2, 256, 256, 1);
  tconv(vf_W1, WT_vf1, 256, 512, 1);
  tconv(vf_W2, WT_vf2, 512, 512, 1);
  tconv(vf_W3, WT_vf3, 512, 16384, 1);
  tconv(attn_Wv, WT_av, 256, 256, 1);
  tconv(attn_Wo, WT_ao, 256, 256, 1);
  tconv(dec_W1, WT_d1, 256, 64, NTSK);
  tconv(dec_W2, WT_d2, 64, 32, NTSK);

  // 2) dt and dX/dt
  {
    int total = (SEQ - 1) * BSZ * DINN;
    prep_dx_kernel<<<(total + 255) / 256, 256, 0, stream>>>(path, tstamp, dts, dX);
  }

  // 3) encoder: x0 @ W1 -> LN -> GELU ; @ W2 -> LN  (x0 rows stride SEQ*DINN)
  gemm_act_kernel<64, 256, true, true, false, true><<<dim3(2, 1), 128, 0, stream>>>(
      path, nullptr, nullptr, 0.f, SEQ * DINN, 0, WT_enc1, enc_b1, enc_g1, enc_be1, henc);
  gemm_act_kernel<256, 256, true, false, true, false><<<dim3(2, 1), 128, 0, stream>>>(
      henc, nullptr, nullptr, 0.f, 256, 0, WT_enc2, enc_b2, enc_g2, enc_be2, zc);

  // 4) RK4 CDE integration
  const float coefs[4] = {0.0f, 0.5f, 0.5f, 1.0f};
  float* ks[4] = {k1, k2, k3, k4};
  for (int s = 0; s < SEQ - 1; ++s) {
    const float* dtp = dts + s;
    const float* dXs = dX + (size_t)s * BSZ * DINN;
    for (int j = 0; j < 4; ++j) {
      const float* kprev = (j == 0) ? nullptr : ks[j - 1];
      gemm_act_kernel<256, 512, true, true, false, true><<<dim3(2, 1), 128, 0, stream>>>(
          zc, kprev, dtp, coefs[j], 256, 0, WT_vf1, vf_b1, vf_g1, vf_be1, h1);
      gemm_act_kernel<512, 512, true, true, true, true><<<dim3(2, 1), 128, 0, stream>>>(
          h1, nullptr, nullptr, 0.f, 512, 0, WT_vf2, vf_b2, vf_g2, vf_be2, h2);
      vf3_contract_kernel<<<dim3(2, 256), 128, 0, stream>>>(h2, WT_vf3, vf_b3, dXs, ks[j]);
    }
    rk4_update_kernel<<<(BSZ * HIDN + 255) / 256, 256, 0, stream>>>(zc, k1, k2, k3, k4, dtp);
  }

  // 5) attention head: v = zf@Wv + bv ; a = v@Wo + bo
  gemm_act_kernel<256, 256, false, false, false, true><<<dim3(2, 1), 128, 0, stream>>>(
      zc, nullptr, nullptr, 0.f, 256, 0, WT_av, attn_bv, nullptr, nullptr, vbuf);
  gemm_act_kernel<256, 256, false, false, true, true><<<dim3(2, 1), 128, 0, stream>>>(
      vbuf, nullptr, nullptr, 0.f, 256, 0, WT_ao, attn_bo, nullptr, nullptr, abuf);

  // 6) per-task decoder MLPs (batched over tasks via grid.y)
  gemm_act_kernel<256, 64, false, true, true, true><<<dim3(2, NTSK), 128, 0, stream>>>(
      abuf, nullptr, nullptr, 0.f, 256, 0, WT_d1, dec_b1, nullptr, nullptr, h1d);
  gemm_act_kernel<64, 32, false, true, true, false><<<dim3(2, NTSK), 128, 0, stream>>>(
      h1d, nullptr, nullptr, 0.f, 64, BSZ * 64, WT_d2, dec_b2, nullptr, nullptr, h2d);
  dec_out_kernel<<<(NTSK * BSZ + 127) / 128, 128, 0, stream>>>(h2d, dec_W3, dec_b3,
                                                               (float*)d_out);
}
